// TemporalBlock_49606872268956
// MI455X (gfx1250) — matrix operand order repaired and checked by reading, not compiled, not run on hardware
//
#include <hip/hip_runtime.h>
#include <hip/hip_bf16.h>

// ---------------- problem constants ----------------
#define B_    8
#define CIN_  128
#define COUT_ 256
#define L_    4096
#define K_    7
#define PAD_  12            // (K-1)*D
#define L2_   4108          // L + PAD
#define L2P_  4112          // L2 padded to multiple of 16
#define XROW_ 4192          // L + 24 front pad + slack (mult of 32)
#define KC1_  896           // CIN*K
#define NT_H  257           // number of 16-wide time tiles covering L2P
#define NT_O  256           // number of 16-wide time tiles covering L

typedef __attribute__((ext_vector_type(16))) _Float16 v16h;
typedef __attribute__((ext_vector_type(8)))  _Float16 v8h;
typedef __attribute__((ext_vector_type(8)))  float    v8f;
typedef __attribute__((ext_vector_type(4)))  int      v4i;

#define WMMA_F32_F16(a, b, c) \
  __builtin_amdgcn_wmma_f32_16x16x32_f16(false, (a), false, (b), (short)0, (c), false, false)

__device__ __forceinline__ v16h cat8(v8h lo, v8h hi) {
  return __builtin_shufflevector(lo, hi, 0, 1, 2, 3, 4, 5, 6, 7, 8, 9, 10, 11, 12, 13, 14, 15);
}

// Exchange 8 f16 with the XOR-16 partner lane (ds_swizzle SWAPX16)
__device__ __forceinline__ v8h swap16(v8h v) {
  v4i u = __builtin_bit_cast(v4i, v);
  v4i r;
#pragma unroll
  for (int i = 0; i < 4; ++i) r[i] = __builtin_amdgcn_ds_swizzle(u[i], 0x401F);
  return __builtin_bit_cast(v8h, r);
}

// A fragment from row-major weights (contiguous K order, matches B fragments):
// lane holds row M=(lane&15), K elems {k0..k0+15}, k0 = 16*(lane>>4)
__device__ __forceinline__ v16h load_a_frag(const _Float16* A, int lda, int lane) {
  const _Float16* p = A + (size_t)(lane & 15) * lda + ((lane >> 4) << 4);
  return cat8(*(const v8h*)p, *(const v8h*)(p + 8));
}

// B fragment from fragment-tiled tensor: lane's 16 f16 (K = 16g+0..15, N = lane&15)
// are stored contiguously (32B per lane slot)
__device__ __forceinline__ v16h load_b_frag_tiled(const _Float16* fragbase, int lane) {
  const _Float16* p = fragbase + lane * 16;
  return cat8(*(const v8h*)p, *(const v8h*)(p + 8));
}

__device__ __forceinline__ const _Float16* frag_ptr(const _Float16* t, int kt, int nt, int NT) {
  return t + ((size_t)kt * NT + nt) * 512;
}
__device__ __forceinline__ _Float16* frag_ptr_w(_Float16* t, int kt, int nt, int NT) {
  return t + ((size_t)kt * NT + nt) * 512;
}

// ---------------- prep kernels ----------------
// x (f32 [B][CIN][L]) -> xpad (f16 [B][CIN][XROW]) with 24 leading zeros
__global__ void k_prep_x(const float* __restrict__ x, _Float16* __restrict__ xp) {
  size_t idx = (size_t)blockIdx.x * blockDim.x + threadIdx.x;
  size_t total = (size_t)B_ * CIN_ * XROW_;
  if (idx >= total) return;
  int    col = (int)(idx % XROW_);
  size_t row = idx / XROW_;
  int l = col - 24;
  float v = (l >= 0 && l < L_) ? x[row * L_ + l] : 0.f;
  xp[idx] = (_Float16)v;
}

// weight-norm: w1h[co][ci*7+k] = g[co] * v / ||v||
__global__ void k_prep_wnorm(const float* __restrict__ v, const float* __restrict__ g,
                             _Float16* __restrict__ w1h) {
  int co = threadIdx.x;
  if (co >= COUT_) return;
  const float* row = v + (size_t)co * KC1_;
  float s = 0.f;
  for (int i = 0; i < KC1_; ++i) s += row[i] * row[i];
  float scale = g[co] * rsqrtf(s);
  _Float16* dst = w1h + (size_t)co * KC1_;
  for (int i = 0; i < KC1_; ++i) dst[i] = (_Float16)(row[i] * scale);
}

__global__ void k_cvt(const float* __restrict__ src, _Float16* __restrict__ dst, int n) {
  int i = blockIdx.x * blockDim.x + threadIdx.x;
  if (i < n) dst[i] = (_Float16)src[i];
}

// im2col fragment tensor for the dilated conv:
// element (b, kt, nt, lane, e): K = kt*32 + 16g + e, (ci,k)=divmod(K,7),
// t = nt*16 + (lane&15); value = xpad[b][ci][t + 2k]
__global__ void k_prep_im2col(const _Float16* __restrict__ xp, _Float16* __restrict__ im) {
  size_t idx = (size_t)blockIdx.x * blockDim.x + threadIdx.x;
  size_t total = (size_t)B_ * 28 * NT_H * 512;
  if (idx >= total) return;
  int e    = (int)(idx & 15);
  int lane = (int)((idx >> 4) & 31);
  size_t s = idx >> 9;
  int nt = (int)(s % NT_H);
  size_t s2 = s / NT_H;
  int kt = (int)(s2 % 28);
  int b  = (int)(s2 / 28);
  int g = lane >> 4;
  int K = kt * 32 + 16 * g + e;
  int ci = K / 7, kk = K - ci * 7;
  int t = nt * 16 + (lane & 15);
  im[idx] = xp[((size_t)b * CIN_ + ci) * XROW_ + t + 2 * kk];
}

// x fragment tensor for the residual GEMM (K dim = ci over 128):
__global__ void k_prep_xfrag(const float* __restrict__ x, _Float16* __restrict__ xB) {
  size_t idx = (size_t)blockIdx.x * blockDim.x + threadIdx.x;
  size_t total = (size_t)B_ * 4 * NT_O * 512;
  if (idx >= total) return;
  int e    = (int)(idx & 15);
  int lane = (int)((idx >> 4) & 31);
  size_t s = idx >> 9;
  int nt = (int)(s & 255);
  size_t s2 = s >> 8;
  int kt = (int)(s2 & 3);
  int b  = (int)(s2 >> 2);
  int g = lane >> 4;
  int ci = kt * 32 + 16 * g + e;
  int t = nt * 16 + (lane & 15);
  xB[idx] = (_Float16)x[((size_t)b * CIN_ + ci) * L_ + t];
}

// ---------------- stage 1: dilated conv GEMM -> hB fragment-tiled ----------------
__global__ void __launch_bounds__(128) k_conv(const _Float16* __restrict__ im,
                                              const _Float16* __restrict__ w1h,
                                              const float* __restrict__ bias,
                                              _Float16* __restrict__ hB) {
  int lane = threadIdx.x & 31, wv = threadIdx.x >> 5;
  int b   = blockIdx.z;
  int co0 = blockIdx.y * 64 + (wv >> 1) * 32;      // M pair base
  int nt0 = blockIdx.x * 4 + (wv & 1) * 2;          // two 16-wide time tiles
  const _Float16* imb = im + (size_t)b * 28 * NT_H * 512;

  v8f acc[2][2] = {};
  for (int kt = 0; kt < 28; ++kt) {
    int k0 = kt * 32;
    v16h a0 = load_a_frag(w1h + (size_t)co0 * KC1_ + k0, KC1_, lane);
    v16h a1 = load_a_frag(w1h + (size_t)(co0 + 16) * KC1_ + k0, KC1_, lane);
    v16h b0 = load_b_frag_tiled(frag_ptr(imb, kt, nt0, NT_H), lane);
    v16h b1 = load_b_frag_tiled(frag_ptr(imb, kt, nt0 + 1, NT_H), lane);
    if (kt + 1 < 28) __builtin_prefetch(frag_ptr(imb, kt + 1, nt0, NT_H) + lane * 16, 0, 1);
    acc[0][0] = WMMA_F32_F16(a0, b0, acc[0][0]);
    acc[0][1] = WMMA_F32_F16(a0, b1, acc[0][1]);
    acc[1][0] = WMMA_F32_F16(a1, b0, acc[1][0]);
    acc[1][1] = WMMA_F32_F16(a1, b1, acc[1][1]);
  }
  int g = lane >> 4, n = lane & 15;
  v8f blo = *(const v8f*)(bias + co0 + 8 * g);
  v8f bhi = *(const v8f*)(bias + co0 + 16 + 8 * g);
  int ktOut = co0 >> 5;
  _Float16* hBb = hB + (size_t)b * 8 * NT_H * 512;
#pragma unroll
  for (int ni = 0; ni < 2; ++ni) {
    int nt = nt0 + ni;
    if (nt >= NT_H) continue;
    int t = nt * 16 + n;
    float keep = (t < PAD_ || t >= L2_) ? 0.f : 1.f;
    v8h h0, h1;                       // own D rows in f16 after epilogue
#pragma unroll
    for (int e = 0; e < 8; ++e) {
      h0[e] = (_Float16)(keep * fmaxf(acc[0][ni][e] + blo[e], 0.f));
      h1[e] = (_Float16)(keep * fmaxf(acc[1][ni][e] + bhi[e], 0.f));
    }
    // cross-half exchange so each lane holds B-frag order K = 16g + 0..15
    v8h send = g ? h0 : h1;
    v8h recv = swap16(send);
    v8h lo = g ? recv : h0;
    v8h hi = g ? h1 : recv;
    _Float16* p = frag_ptr_w(hBb, ktOut, nt, NT_H) + lane * 16;
    *(v8h*)p = lo;
    *(v8h*)(p + 8) = hi;
  }
}

// ---------------- stage 2: QKV GEMM -> qkvT plain [B][768][L2P] f16 ----------------
__global__ void __launch_bounds__(128) k_qkv(const _Float16* __restrict__ hB,
                                             const _Float16* __restrict__ qw,
                                             const float* __restrict__ qb,
                                             _Float16* __restrict__ qkvT) {
  int lane = threadIdx.x & 31, wv = threadIdx.x >> 5;
  int b  = blockIdx.z;
  int j0 = blockIdx.y * 64 + (wv >> 1) * 32;
  int nt0 = blockIdx.x * 4 + (wv & 1) * 2;
  const _Float16* hBb = hB + (size_t)b * 8 * NT_H * 512;

  v8f acc[2][2] = {};
  for (int kt = 0; kt < 8; ++kt) {
    int k0 = kt * 32;
    v16h a0 = load_a_frag(qw + (size_t)j0 * COUT_ + k0, COUT_, lane);
    v16h a1 = load_a_frag(qw + (size_t)(j0 + 16) * COUT_ + k0, COUT_, lane);
    v16h b0 = load_b_frag_tiled(frag_ptr(hBb, kt, nt0, NT_H), lane);
    v16h b1 = load_b_frag_tiled(frag_ptr(hBb, kt, nt0 + 1, NT_H), lane);
    if (kt + 1 < 8) __builtin_prefetch(frag_ptr(hBb, kt + 1, nt0, NT_H) + lane * 16, 0, 1);
    acc[0][0] = WMMA_F32_F16(a0, b0, acc[0][0]);
    acc[0][1] = WMMA_F32_F16(a0, b1, acc[0][1]);
    acc[1][0] = WMMA_F32_F16(a1, b0, acc[1][0]);
    acc[1][1] = WMMA_F32_F16(a1, b1, acc[1][1]);
  }
  int g = lane >> 4, n = lane & 15;
#pragma unroll
  for (int mi = 0; mi < 2; ++mi) {
    v8f bj = *(const v8f*)(qb + j0 + mi * 16 + 8 * g);
#pragma unroll
    for (int ni = 0; ni < 2; ++ni) {
      int nt = nt0 + ni;
      if (nt >= NT_H) continue;
      int t = nt * 16 + n;
#pragma unroll
      for (int r = 0; r < 8; ++r) {
        int j = j0 + mi * 16 + r + 8 * g;
        qkvT[((size_t)b * (3 * COUT_) + j) * L2P_ + t] = (_Float16)(acc[mi][ni][r] + bj[r]);
      }
    }
  }
}

// ---------------- stage 3: NATTEN attention -> oB fragment-tiled ----------------
__global__ void __launch_bounds__(128) k_attn(const _Float16* __restrict__ qkvT,
                                              const float* __restrict__ rpb,
                                              _Float16* __restrict__ oB) {
  __shared__ _Float16 sm[4 * 2 * 32 * 16];      // per wave: 2 fragments of 512 f16
  int lane = threadIdx.x & 31, wv = threadIdx.x >> 5;
  int b  = blockIdx.y;
  int t0 = blockIdx.x * 128 + wv * 32;
  int t  = t0 + lane;

  const _Float16* qp = qkvT + (size_t)b * (3 * COUT_) * L2P_;
  const _Float16* kp = qp + (size_t)COUT_ * L2P_;
  const _Float16* vp = qp + (size_t)2 * COUT_ * L2P_;

  int r  = t & 1, ii = t >> 1;
  int Lg = (L2_ - r + 1) >> 1;
  int hi = Lg - K_;
  int start = ii - (K_ / 2);
  start = start < 0 ? 0 : (start > hi ? hi : start);

  int   pos[K_];
  float sc[K_];
#pragma unroll
  for (int kk = 0; kk < K_; ++kk) {
    pos[kk] = r + 2 * (start + kk);
    sc[kk]  = rpb[start + kk - ii + (K_ - 1)];
  }
  // scores
  for (int c = 0; c < COUT_; ++c) {
    float qv = (float)qp[(size_t)c * L2P_ + t] * 0.0625f;   // COUT^-0.5
#pragma unroll
    for (int kk = 0; kk < K_; ++kk)
      sc[kk] += qv * (float)kp[(size_t)c * L2P_ + pos[kk]];
  }
  // softmax over 7
  float mx = sc[0];
#pragma unroll
  for (int kk = 1; kk < K_; ++kk) mx = fmaxf(mx, sc[kk]);
  float se = 0.f;
#pragma unroll
  for (int kk = 0; kk < K_; ++kk) { sc[kk] = __expf(sc[kk] - mx); se += sc[kk]; }
  float inv = 1.f / se;
#pragma unroll
  for (int kk = 0; kk < K_; ++kk) sc[kk] *= inv;

  // weighted sum of V, staged into fragment layout through LDS.
  // B-frag convention: slot = (t&15) + 16*g, element e=c&15, g=(c>>4)&1, kt=c>>5
  _Float16* smw = sm + wv * 2 * 512;
  int f = lane >> 4;                 // which 16-wide time tile this t belongs to
  int ntBase = t0 >> 4;
  _Float16* oBb = oB + (size_t)b * 8 * NT_O * 512;

  for (int kt = 0; kt < 8; ++kt) {
#pragma unroll
    for (int g = 0; g < 2; ++g) {
      int cbase = kt * 32 + g * 16;
      v8h eA, eB;
#pragma unroll
      for (int j = 0; j < 8; ++j) {
        int c0 = cbase + j, c1 = cbase + 8 + j;
        float o0 = 0.f, o1 = 0.f;
#pragma unroll
        for (int kk = 0; kk < K_; ++kk) {
          o0 += sc[kk] * (float)vp[(size_t)c0 * L2P_ + pos[kk]];
          o1 += sc[kk] * (float)vp[(size_t)c1 * L2P_ + pos[kk]];
        }
        eA[j] = (_Float16)o0;
        eB[j] = (_Float16)o1;
      }
      int slot = (lane & 15) + 16 * g;
      _Float16* p = smw + ((size_t)f * 32 + slot) * 16;
      *(v8h*)p = eA;
      *(v8h*)(p + 8) = eB;
    }
    // stream the two completed fragments for this kt out to global
#pragma unroll
    for (int ff = 0; ff < 2; ++ff) {
      const _Float16* p = smw + ((size_t)ff * 32 + lane) * 16;
      v8h lo = *(const v8h*)p;
      v8h hh = *(const v8h*)(p + 8);
      _Float16* q = frag_ptr_w(oBb, kt, ntBase + ff, NT_O) + lane * 16;
      *(v8h*)q = lo;
      *(v8h*)(q + 8) = hh;
    }
  }
}

// ---------------- stage 4: proj GEMM + relu + residual GEMM + relu -> out f32 ----------------
__global__ void __launch_bounds__(128) k_projres(const _Float16* __restrict__ oB,
                                                 const _Float16* __restrict__ pw,
                                                 const float* __restrict__ pb,
                                                 const _Float16* __restrict__ dw,
                                                 const float* __restrict__ db,
                                                 const _Float16* __restrict__ xB,
                                                 float* __restrict__ out) {
  int lane = threadIdx.x & 31, wv = threadIdx.x >> 5;
  int b   = blockIdx.z;
  int co0 = blockIdx.y * 64 + (wv >> 1) * 32;
  int nt0 = blockIdx.x * 4 + (wv & 1) * 2;
  const _Float16* oBb = oB + (size_t)b * 8 * NT_O * 512;
  const _Float16* xBb = xB + (size_t)b * 4 * NT_O * 512;
  int g = lane >> 4, n = lane & 15;

  v8f acc[2][2] = {};
  // phase 1: projection over attention output (K = 256)
  for (int kt = 0; kt < 8; ++kt) {
    int k0 = kt * 32;
    v16h a0 = load_a_frag(pw + (size_t)co0 * COUT_ + k0, COUT_, lane);
    v16h a1 = load_a_frag(pw + (size_t)(co0 + 16) * COUT_ + k0, COUT_, lane);
    v16h b0 = load_b_frag_tiled(frag_ptr(oBb, kt, nt0, NT_O), lane);
    v16h b1 = load_b_frag_tiled(frag_ptr(oBb, kt, nt0 + 1, NT_O), lane);
    if (kt + 1 < 8) __builtin_prefetch(frag_ptr(oBb, kt + 1, nt0, NT_O) + lane * 16, 0, 1);
    acc[0][0] = WMMA_F32_F16(a0, b0, acc[0][0]);
    acc[0][1] = WMMA_F32_F16(a0, b1, acc[0][1]);
    acc[1][0] = WMMA_F32_F16(a1, b0, acc[1][0]);
    acc[1][1] = WMMA_F32_F16(a1, b1, acc[1][1]);
  }
  // + proj bias, first relu
  {
    v8f p0 = *(const v8f*)(pb + co0 + 8 * g);
    v8f p1 = *(const v8f*)(pb + co0 + 16 + 8 * g);
#pragma unroll
    for (int ni = 0; ni < 2; ++ni)
#pragma unroll
      for (int e = 0; e < 8; ++e) {
        acc[0][ni][e] = fmaxf(acc[0][ni][e] + p0[e], 0.f);
        acc[1][ni][e] = fmaxf(acc[1][ni][e] + p1[e], 0.f);
      }
  }
  // phase 2: downsample residual over x (K = 128)
  for (int kt = 0; kt < 4; ++kt) {
    int k0 = kt * 32;
    v16h a0 = load_a_frag(dw + (size_t)co0 * CIN_ + k0, CIN_, lane);
    v16h a1 = load_a_frag(dw + (size_t)(co0 + 16) * CIN_ + k0, CIN_, lane);
    v16h b0 = load_b_frag_tiled(frag_ptr(xBb, kt, nt0, NT_O), lane);
    v16h b1 = load_b_frag_tiled(frag_ptr(xBb, kt, nt0 + 1, NT_O), lane);
    acc[0][0] = WMMA_F32_F16(a0, b0, acc[0][0]);
    acc[0][1] = WMMA_F32_F16(a0, b1, acc[0][1]);
    acc[1][0] = WMMA_F32_F16(a1, b0, acc[1][0]);
    acc[1][1] = WMMA_F32_F16(a1, b1, acc[1][1]);
  }
  // + ds bias, final relu, store fp32
  v8f d0 = *(const v8f*)(db + co0 + 8 * g);
  v8f d1 = *(const v8f*)(db + co0 + 16 + 8 * g);
#pragma unroll
  for (int mi = 0; mi < 2; ++mi)
#pragma unroll
    for (int ni = 0; ni < 2; ++ni) {
      int t = (nt0 + ni) * 16 + n;
#pragma unroll
      for (int e = 0; e < 8; ++e) {
        int co = co0 + mi * 16 + e + 8 * g;
        float bias = (mi == 0) ? d0[e] : d1[e];
        float v = fmaxf(acc[mi][ni][e] + bias, 0.f);
        out[((size_t)b * COUT_ + co) * L_ + t] = v;
      }
    }
}

// ---------------- launcher ----------------
extern "C" void kernel_launch(void* const* d_in, const int* in_sizes, int n_in,
                              void* d_out, int out_size, void* d_ws, size_t ws_size,
                              hipStream_t stream) {
  (void)in_sizes; (void)n_in; (void)out_size; (void)ws_size;
  const float* x       = (const float*)d_in[0];
  const float* conv1_v = (const float*)d_in[1];
  const float* conv1_g = (const float*)d_in[2];
  const float* conv1_b = (const float*)d_in[3];
  const float* qkv_w   = (const float*)d_in[4];
  const float* qkv_b   = (const float*)d_in[5];
  const float* rpb     = (const float*)d_in[6];
  const float* proj_w  = (const float*)d_in[7];
  const float* proj_b  = (const float*)d_in[8];
  const float* ds_w    = (const float*)d_in[9];
  const float* ds_b    = (const float*)d_in[10];
  float* out = (float*)d_out;

  size_t off = 0;
  char* base = (char*)d_ws;
  auto take = [&](size_t elems) {
    _Float16* p = (_Float16*)(base + off);
    off += (elems * sizeof(_Float16) + 255) & ~(size_t)255;
    return p;
  };
  _Float16* xpad   = take((size_t)B_ * CIN_ * XROW_);
  _Float16* w1h    = take((size_t)COUT_ * KC1_);
  _Float16* qkvwh  = take((size_t)3 * COUT_ * COUT_);
  _Float16* projwh = take((size_t)COUT_ * COUT_);
  _Float16* dswh   = take((size_t)COUT_ * CIN_);
  _Float16* im2col = take((size_t)B_ * 28 * NT_H * 512);
  _Float16* hB     = take((size_t)B_ * 8 * NT_H * 512);
  _Float16* qkvT   = take((size_t)B_ * 3 * COUT_ * L2P_);
  _Float16* oB     = take((size_t)B_ * 8 * NT_O * 512);
  _Float16* xB     = take((size_t)B_ * 4 * NT_O * 512);

  {
    size_t ntot = (size_t)B_ * CIN_ * XROW_;
    k_prep_x<<<dim3((unsigned)((ntot + 255) / 256)), 256, 0, stream>>>(x, xpad);
  }
  k_prep_wnorm<<<1, 256, 0, stream>>>(conv1_v, conv1_g, w1h);
  k_cvt<<<(3 * COUT_ * COUT_ + 255) / 256, 256, 0, stream>>>(qkv_w, qkvwh, 3 * COUT_ * COUT_);
  k_cvt<<<(COUT_ * COUT_ + 255) / 256, 256, 0, stream>>>(proj_w, projwh, COUT_ * COUT_);
  k_cvt<<<(COUT_ * CIN_ + 255) / 256, 256, 0, stream>>>(ds_w, dswh, COUT_ * CIN_);
  {
    size_t ntot = (size_t)B_ * 28 * NT_H * 512;
    k_prep_im2col<<<dim3((unsigned)((ntot + 255) / 256)), 256, 0, stream>>>(xpad, im2col);
  }
  {
    size_t ntot = (size_t)B_ * 4 * NT_O * 512;
    k_prep_xfrag<<<dim3((unsigned)((ntot + 255) / 256)), 256, 0, stream>>>(x, xB);
  }

  // GEMM grids: block = 4 waves = 64M x 64t per block
  k_conv<<<dim3(65, 4, B_), 128, 0, stream>>>(im2col, w1h, conv1_b, hB);
  k_qkv <<<dim3(65, 12, B_), 128, 0, stream>>>(hB, qkvwh, qkv_b, qkvT);
  k_attn<<<dim3(L_ / 128, B_), 128, 0, stream>>>(qkvT, rpb, oB);
  k_projres<<<dim3(64, 4, B_), 128, 0, stream>>>(oB, projwh, proj_b, dswh, ds_b, xB, out);
}
